// HGCML_2662879724016
// MI455X (gfx1250) — matrix operand order repaired and checked by reading, not compiled, not run on hardware
//
#include <hip/hip_runtime.h>
#include <hip/hip_bf16.h>
#include <math.h>

typedef __attribute__((ext_vector_type(8)))  float  v8f;
typedef __attribute__((ext_vector_type(16))) __bf16 v16bf;

#define TAU_INV 5.0f
#define NROWS 4096
#define DDIM  512
#define BM 128
#define BN 128
#define BK 32
#define LDT 20   // dword stride of LDS tile rows: 16 data dwords + 4 pad
                 // rows stay 16B-aligned; r*20 mod 64 hits 16 distinct banks

union Frag16 { unsigned int u[8]; v16bf v; };

__device__ __forceinline__ unsigned short f2bf(float f) {
    union { float f; unsigned int u; } x; x.f = f;
    unsigned int u = x.u;
    unsigned int r = u + 0x7FFFu + ((u >> 16) & 1u);  // round-to-nearest-even
    return (unsigned short)(r >> 16);
}

// ---------------------------------------------------------------------------
// Fused bf16 WMMA GEMM: D = A[M,K] x B[Ncols,K]^T  (both row-major, K contig)
// Double-buffered LDS staging (b128 loads/stores), 8 waves x (2x4) 16x16x32
// WMMA accumulators per 128x128 block tile.
// MODE 0: out_bf = bf16(prelu(acc + bias[col]))
// MODE 1: out_f  = acc + bias[col]
// MODE 2: e = exp(acc/tau); accumulate row sums / masked row sums
//         (COLS: also column sums / masked column sums, same mask element)
// ---------------------------------------------------------------------------
template<int MODE, bool MASK_T, bool COLS>
__global__ __launch_bounds__(256)
void gemm_bf16_wmma(const unsigned short* __restrict__ A,
                    const unsigned short* __restrict__ Bm,
                    int M, int Ncols, int K,
                    const float* __restrict__ bias,
                    const float* __restrict__ prelu_a,
                    unsigned short* __restrict__ out_bf,
                    float* __restrict__ out_f,
                    const int* __restrict__ mask, int mld,
                    float* __restrict__ rowsum, float* __restrict__ rowm,
                    float* __restrict__ colsum, float* __restrict__ colm)
{
    __shared__ __align__(16) unsigned int As[2][BM * LDT];
    __shared__ __align__(16) unsigned int Bs[2][BN * LDT];
    __shared__ float rAcc[BM], rmAcc[BM], cAcc[BN], cmAcc[BN];

    const int tid  = threadIdx.x;
    const int lane = tid & 31;
    const int wave = tid >> 5;
    const int hf   = lane >> 4;     // half-wave (0/1)
    const int l16  = lane & 15;
    const int wm   = (wave >> 1) * 32;   // wave tile row offset in block
    const int wn   = (wave & 1) * 64;    // wave tile col offset in block
    const int bm   = blockIdx.y * BM;
    const int bn   = blockIdx.x * BN;

    if (MODE == 2) {
        for (int i = tid; i < BM; i += 256) { rAcc[i] = 0.f; rmAcc[i] = 0.f; }
        if (COLS)
            for (int i = tid; i < BN; i += 256) { cAcc[i] = 0.f; cmAcc[i] = 0.f; }
    }

    v8f acc[2][4];
    #pragma unroll
    for (int mt = 0; mt < 2; mt++)
        #pragma unroll
        for (int nt = 0; nt < 4; nt++)
            acc[mt][nt] = (v8f){0.f,0.f,0.f,0.f,0.f,0.f,0.f,0.f};

    const uint4* Ag = (const uint4*)A;
    const uint4* Bg = (const uint4*)Bm;
    const int Kq = K >> 3;   // uint4 per row

    // staging assignment: 512 uint4 per tile, 2 per thread per matrix
    const int q0 = tid,        r0 = q0 >> 2, c0 = q0 & 3;
    const int q1 = tid + 256,  r1 = q1 >> 2, c1 = q1 & 3;

    uint4 ra0, ra1, rb0, rb1;

    // prologue: stage step 0 into buffer 0
    ra0 = Ag[(size_t)(bm + r0) * Kq + c0];
    ra1 = Ag[(size_t)(bm + r1) * Kq + c1];
    rb0 = Bg[(size_t)(bn + r0) * Kq + c0];
    rb1 = Bg[(size_t)(bn + r1) * Kq + c1];
    *(uint4*)&As[0][r0 * LDT + c0 * 4] = ra0;
    *(uint4*)&As[0][r1 * LDT + c1 * 4] = ra1;
    *(uint4*)&Bs[0][r0 * LDT + c0 * 4] = rb0;
    *(uint4*)&Bs[0][r1 * LDT + c1 * 4] = rb1;
    __syncthreads();

    const int nsteps = K / BK;
    for (int s = 0; s < nsteps; s++) {
        const int  cur  = s & 1;
        const bool more = (s + 1) < nsteps;
        if (more) {
            const int o = (s + 1) * 4;   // uint4 offset within row for next step
            ra0 = Ag[(size_t)(bm + r0) * Kq + o + c0];
            ra1 = Ag[(size_t)(bm + r1) * Kq + o + c1];
            rb0 = Bg[(size_t)(bn + r0) * Kq + o + c0];
            rb1 = Bg[(size_t)(bn + r1) * Kq + o + c1];
        }

        // B fragments: 32x16, dwords (v&3)+hf*4+(v>>2)*8 -> two aligned quads
        Frag16 bfr[4];
        #pragma unroll
        for (int nt = 0; nt < 4; nt++) {
            const int nb = (wn + nt * 16 + l16) * LDT + hf * 4;
            *(uint4*)&bfr[nt].u[0] = *(const uint4*)&Bs[cur][nb];
            *(uint4*)&bfr[nt].u[4] = *(const uint4*)&Bs[cur][nb + 8];
        }
        #pragma unroll
        for (int mt = 0; mt < 2; mt++) {
            // A fragment: 16x32, dwords (v&3)+hf*4+(v>>2)*8 -> two aligned quads
            Frag16 af;
            const int ab = (wm + mt * 16 + l16) * LDT + hf * 4;
            *(uint4*)&af.u[0] = *(const uint4*)&As[cur][ab];
            *(uint4*)&af.u[4] = *(const uint4*)&As[cur][ab + 8];
            #pragma unroll
            for (int nt = 0; nt < 4; nt++)
                acc[mt][nt] = __builtin_amdgcn_wmma_f32_16x16x32_bf16(
                    false, af.v, false, bfr[nt].v, (short)0, acc[mt][nt],
                    false, false);
        }

        if (more) {
            const int nb = 1 - cur;
            *(uint4*)&As[nb][r0 * LDT + c0 * 4] = ra0;
            *(uint4*)&As[nb][r1 * LDT + c1 * 4] = ra1;
            *(uint4*)&Bs[nb][r0 * LDT + c0 * 4] = rb0;
            *(uint4*)&Bs[nb][r1 * LDT + c1 * 4] = rb1;
            __syncthreads();
        }
    }

    if (MODE == 0 || MODE == 1) {
        const float pa = (MODE == 0) ? prelu_a[0] : 0.f;
        #pragma unroll
        for (int mt = 0; mt < 2; mt++)
            #pragma unroll
            for (int v = 0; v < 8; v++) {
                int row = bm + wm + mt * 16 + v + 8 * hf;
                #pragma unroll
                for (int nt = 0; nt < 4; nt++) {
                    int col = bn + wn + nt * 16 + l16;
                    float val = acc[mt][nt][v] + bias[col];
                    if (MODE == 0) {
                        val = (val >= 0.f) ? val : pa * val;
                        out_bf[(size_t)row * Ncols + col] = f2bf(val);
                    } else {
                        out_f[(size_t)row * Ncols + col] = val;
                    }
                }
            }
    } else {
        float ec[4]  = {0.f, 0.f, 0.f, 0.f};
        float ecm[4] = {0.f, 0.f, 0.f, 0.f};
        #pragma unroll
        for (int mt = 0; mt < 2; mt++) {
            #pragma unroll
            for (int v = 0; v < 8; v++) {
                int row = bm + wm + mt * 16 + v + 8 * hf;
                float er = 0.f, erm = 0.f;
                #pragma unroll
                for (int nt = 0; nt < 4; nt++) {
                    int col = bn + wn + nt * 16 + l16;
                    float e = __expf(acc[mt][nt][v] * TAU_INV);
                    float mk = (float)(MASK_T ? mask[(size_t)col * mld + row]
                                              : mask[(size_t)row * mld + col]);
                    er  += e;
                    erm += e * mk;
                    if (COLS) { ec[nt] += e; ecm[nt] += e * mk; }
                }
                // reduce across the 16 lanes that share this row
                #pragma unroll
                for (int off = 1; off < 16; off <<= 1) {
                    er  += __shfl_xor(er,  off, 32);
                    erm += __shfl_xor(erm, off, 32);
                }
                if (l16 == 0) {
                    int ridx = wm + mt * 16 + v + 8 * hf;
                    atomicAdd(&rAcc[ridx],  er);
                    atomicAdd(&rmAcc[ridx], erm);
                }
            }
        }
        if (COLS) {
            #pragma unroll
            for (int nt = 0; nt < 4; nt++) {
                float c0s = ec[nt]  + __shfl_xor(ec[nt],  16, 32);
                float c1s = ecm[nt] + __shfl_xor(ecm[nt], 16, 32);
                if (hf == 0) {
                    int cidx = wn + nt * 16 + l16;
                    atomicAdd(&cAcc[cidx],  c0s);
                    atomicAdd(&cmAcc[cidx], c1s);
                }
            }
        }
        __syncthreads();
        if (tid < BM) {
            atomicAdd(&rowsum[bm + tid], rAcc[tid]);
            atomicAdd(&rowm[bm + tid],  rmAcc[tid]);
        }
        if (COLS && tid < BN) {
            atomicAdd(&colsum[bn + tid], cAcc[tid]);
            atomicAdd(&colm[bn + tid],  cmAcc[tid]);
        }
    }
}

// ---------------------------------------------------------------------------
__global__ void cvt_f32_bf16(const float* __restrict__ in,
                             unsigned short* __restrict__ out, int n) {
    int i = blockIdx.x * blockDim.x + threadIdx.x;
    if (i < n) out[i] = f2bf(in[i]);
}

// row-wise L2 normalize [NROWS, DDIM] f32 -> bf16
__global__ __launch_bounds__(256)
void norm_rows(const float* __restrict__ h, unsigned short* __restrict__ out) {
    __shared__ float red[8];
    int row = blockIdx.x;
    const float* hr = h + (size_t)row * DDIM;
    float ss = 0.f;
    for (int k = threadIdx.x; k < DDIM; k += 256) { float v = hr[k]; ss += v * v; }
    for (int off = 16; off; off >>= 1) ss += __shfl_xor(ss, off, 32);
    int lane = threadIdx.x & 31, w = threadIdx.x >> 5;
    if (lane == 0) red[w] = ss;
    __syncthreads();
    if (threadIdx.x == 0) {
        float t = 0.f;
        for (int i = 0; i < 8; i++) t += red[i];
        red[0] = t;
    }
    __syncthreads();
    float inv = 1.f / fmaxf(sqrtf(red[0]), 1e-12f);
    for (int k = threadIdx.x; k < DDIM; k += 256)
        out[(size_t)row * DDIM + k] = f2bf(hr[k] * inv);
}

// column mean of z [NROWS, DDIM] -> s [DDIM]
__global__ void colmean(const float* __restrict__ z, float* __restrict__ s) {
    int d = blockIdx.x * blockDim.x + threadIdx.x;
    if (d < DDIM) {
        float acc = 0.f;
        for (int n = 0; n < NROWS; n++) acc += z[(size_t)n * DDIM + d];
        s[d] = acc * (1.f / (float)NROWS);
    }
}

// tiny global projector + bilinear: summ = W @ proj(s) ; one block, f32
__global__ __launch_bounds__(256)
void gproj(const float* __restrict__ s,
           const float* __restrict__ gw1, const float* __restrict__ gb1,
           const float* __restrict__ ga,
           const float* __restrict__ gw2, const float* __restrict__ gb2,
           const float* __restrict__ W, float* __restrict__ summ) {
    __shared__ float sv[DDIM], tv[DDIM], hv[DDIM];
    for (int i = threadIdx.x; i < DDIM; i += 256) sv[i] = s[i];
    __syncthreads();
    float a = ga[0];
    for (int d = threadIdx.x; d < DDIM; d += 256) {
        float acc = gb1[d];
        const float* wr = gw1 + (size_t)d * DDIM;
        for (int k = 0; k < DDIM; k++) acc += wr[k] * sv[k];
        tv[d] = (acc >= 0.f) ? acc : a * acc;
    }
    __syncthreads();
    for (int d = threadIdx.x; d < DDIM; d += 256) {
        float acc = gb2[d];
        const float* wr = gw2 + (size_t)d * DDIM;
        for (int k = 0; k < DDIM; k++) acc += wr[k] * tv[k];
        hv[d] = acc;
    }
    __syncthreads();
    for (int d = threadIdx.x; d < DDIM; d += 256) {
        float acc = 0.f;
        const float* wr = W + (size_t)d * DDIM;
        for (int k = 0; k < DDIM; k++) acc += wr[k] * hv[k];
        summ[d] = acc;
    }
}

// DGI term: accumulate sum_i -log(sig(z_i . summ)+eps) (pos) or -log(1-sig+eps)
__global__ __launch_bounds__(256)
void dgi_loss(const float* __restrict__ z, const float* __restrict__ summ,
              int positive, float* __restrict__ acc) {
    int lane = threadIdx.x & 31, w = threadIdx.x >> 5;
    int row = blockIdx.x * 8 + w;
    const float* zr = z + (size_t)row * DDIM;
    float p = 0.f;
    for (int k = lane; k < DDIM; k += 32) p += zr[k] * summ[k];
    for (int off = 16; off; off >>= 1) p += __shfl_xor(p, off, 32);
    if (lane == 0) {
        float d = 1.f / (1.f + __expf(-p));
        float term = positive ? -logf(d + 1e-15f) : -logf(1.f - d + 1e-15f);
        atomicAdd(acc, term);
    }
}

// per-row InfoNCE log terms -> scalar accumulator
__global__ void local_loss(const float* __restrict__ rsaa, const float* __restrict__ maa,
                           const float* __restrict__ rsab, const float* __restrict__ mab,
                           const float* __restrict__ rsbb, const float* __restrict__ mbb,
                           const float* __restrict__ rsba, const float* __restrict__ mba,
                           float* __restrict__ acc) {
    int i = blockIdx.x * blockDim.x + threadIdx.x;
    if (i < NROWS) {
        float t1 = -logf(mab[i] / (rsaa[i] + rsab[i] - maa[i]));
        float t2 = -logf(mba[i] / (rsbb[i] + rsba[i] - mbb[i]));
        atomicAdd(acc, t1 + t2);
    }
}

__global__ void final_combine(const float* __restrict__ localAcc,
                              const float* __restrict__ gAcc,
                              float* __restrict__ out) {
    float local = 0.5f * localAcc[0] / (float)NROWS;      // 0.5*(L1+L2), each a mean
    float glob  = 0.25f * gAcc[0] / (float)NROWS;         // 0.5*avg of 2x 0.5*(pos+neg)
    out[0] = 0.5f * local + 0.5f * glob;                  // ALPHA = 0.5
}

// ---------------------------------------------------------------------------
extern "C" void kernel_launch(void* const* d_in, const int* in_sizes, int n_in,
                              void* d_out, int out_size, void* d_ws, size_t ws_size,
                              hipStream_t stream) {
    const float* z1  = (const float*)d_in[0];
    const float* z2  = (const float*)d_in[1];
    const float* lw1 = (const float*)d_in[2];
    const float* lb1 = (const float*)d_in[3];
    const float* la  = (const float*)d_in[4];
    const float* lw2 = (const float*)d_in[5];
    const float* lb2 = (const float*)d_in[6];
    const float* gw1 = (const float*)d_in[7];
    const float* gb1 = (const float*)d_in[8];
    const float* ga  = (const float*)d_in[9];
    const float* gw2 = (const float*)d_in[10];
    const float* gb2 = (const float*)d_in[11];
    const float* W   = (const float*)d_in[12];
    const int*   mask= (const int*)d_in[13];
    float* out = (float*)d_out;

    // workspace carve-up
    char* p = (char*)d_ws;
    const size_t ND = (size_t)NROWS * DDIM;           // 2,097,152
    const size_t DD = (size_t)DDIM * DDIM;            // 262,144
    unsigned short* lw1b   = (unsigned short*)p;  p += DD * 2;
    unsigned short* lw2b   = (unsigned short*)p;  p += DD * 2;
    unsigned short* a_norm = (unsigned short*)p;  p += ND * 2;
    unsigned short* b_norm = (unsigned short*)p;  p += ND * 2;
    unsigned short* zb     = (unsigned short*)p;  p += ND * 2;
    unsigned short* actb   = (unsigned short*)p;  p += ND * 2;
    float* hf32 = (float*)p;                      p += ND * 4;
    float* zero_base = (float*)p;                 // zeroed region starts here
    float* rs_aa = (float*)p; p += NROWS * 4;
    float* m_aa  = (float*)p; p += NROWS * 4;
    float* rs_ab = (float*)p; p += NROWS * 4;
    float* m_ab  = (float*)p; p += NROWS * 4;
    float* rs_bb = (float*)p; p += NROWS * 4;
    float* m_bb  = (float*)p; p += NROWS * 4;
    float* rs_ba = (float*)p; p += NROWS * 4;
    float* m_ba  = (float*)p; p += NROWS * 4;
    float* localAcc = (float*)p; p += 4;
    float* gAcc     = (float*)p; p += 4;
    size_t zero_bytes = (size_t)((char*)p - (char*)zero_base);
    float* s1    = (float*)p; p += DDIM * 4;
    float* s2    = (float*)p; p += DDIM * 4;
    float* summ1 = (float*)p; p += DDIM * 4;
    float* summ2 = (float*)p; p += DDIM * 4;

    hipMemsetAsync(zero_base, 0, zero_bytes, stream);

    // weights -> bf16
    cvt_f32_bf16<<<(DD + 255) / 256, 256, 0, stream>>>(lw1, lw1b, (int)DD);
    cvt_f32_bf16<<<(DD + 255) / 256, 256, 0, stream>>>(lw2, lw2b, (int)DD);

    dim3 blk(256);
    dim3 grid_proj(DDIM / BN, NROWS / BM);   // 4 x 32
    dim3 grid_sim(NROWS / BN, NROWS / BM);   // 32 x 32

    // ---- projector: z1 -> a_norm
    cvt_f32_bf16<<<(ND + 255) / 256, 256, 0, stream>>>(z1, zb, (int)ND);
    gemm_bf16_wmma<0, false, false><<<grid_proj, blk, 0, stream>>>(
        zb, lw1b, NROWS, DDIM, DDIM, lb1, la, actb, nullptr,
        nullptr, 0, nullptr, nullptr, nullptr, nullptr);
    gemm_bf16_wmma<1, false, false><<<grid_proj, blk, 0, stream>>>(
        actb, lw2b, NROWS, DDIM, DDIM, lb2, nullptr, nullptr, hf32,
        nullptr, 0, nullptr, nullptr, nullptr, nullptr);
    norm_rows<<<NROWS, 256, 0, stream>>>(hf32, a_norm);

    // ---- projector: z2 -> b_norm
    cvt_f32_bf16<<<(ND + 255) / 256, 256, 0, stream>>>(z2, zb, (int)ND);
    gemm_bf16_wmma<0, false, false><<<grid_proj, blk, 0, stream>>>(
        zb, lw1b, NROWS, DDIM, DDIM, lb1, la, actb, nullptr,
        nullptr, 0, nullptr, nullptr, nullptr, nullptr);
    gemm_bf16_wmma<1, false, false><<<grid_proj, blk, 0, stream>>>(
        actb, lw2b, NROWS, DDIM, DDIM, lb2, nullptr, nullptr, hf32,
        nullptr, 0, nullptr, nullptr, nullptr, nullptr);
    norm_rows<<<NROWS, 256, 0, stream>>>(hf32, b_norm);

    // ---- similarities (fused exp + masked row/col sums)
    gemm_bf16_wmma<2, false, false><<<grid_sim, blk, 0, stream>>>(
        a_norm, a_norm, NROWS, NROWS, DDIM, nullptr, nullptr, nullptr, nullptr,
        mask, NROWS, rs_aa, m_aa, nullptr, nullptr);
    gemm_bf16_wmma<2, true, false><<<grid_sim, blk, 0, stream>>>(
        b_norm, b_norm, NROWS, NROWS, DDIM, nullptr, nullptr, nullptr, nullptr,
        mask, NROWS, rs_bb, m_bb, nullptr, nullptr);
    gemm_bf16_wmma<2, false, true><<<grid_sim, blk, 0, stream>>>(
        a_norm, b_norm, NROWS, NROWS, DDIM, nullptr, nullptr, nullptr, nullptr,
        mask, NROWS, rs_ab, m_ab, rs_ba, m_ba);

    local_loss<<<NROWS / 256, 256, 0, stream>>>(
        rs_aa, m_aa, rs_ab, m_ab, rs_bb, m_bb, rs_ba, m_ba, localAcc);

    // ---- global (DGI) branch, f32 (tiny)
    colmean<<<2, 256, 0, stream>>>(z1, s1);
    colmean<<<2, 256, 0, stream>>>(z2, s2);
    gproj<<<1, 256, 0, stream>>>(s1, gw1, gb1, ga, gw2, gb2, W, summ1);
    gproj<<<1, 256, 0, stream>>>(s2, gw1, gb1, ga, gw2, gb2, W, summ2);
    dgi_loss<<<NROWS / 8, 256, 0, stream>>>(z1, summ1, 1, gAcc);
    dgi_loss<<<NROWS / 8, 256, 0, stream>>>(z2, summ1, 0, gAcc);
    dgi_loss<<<NROWS / 8, 256, 0, stream>>>(z2, summ2, 1, gAcc);
    dgi_loss<<<NROWS / 8, 256, 0, stream>>>(z1, summ2, 0, gAcc);

    final_combine<<<1, 1, 0, stream>>>(localAcc, gAcc, out);
}
